// LayerNormLSTM_54348516164102
// MI455X (gfx1250) — matrix operand order repaired and checked by reading, not compiled, not run on hardware
//
#include <hip/hip_runtime.h>
#include <hip/hip_bf16.h>

// ---------------------------------------------------------------------------
// LayerNorm LSTM cell, fused for MI455X (gfx1250, wave32, WMMA).
//   B = 8192, IN = H = 1024, gates = 4H = 4096
// Strategy:
//   kernel 1: f32 -> bf16 conversion of input, h_prev, w_i, w_h into d_ws
//   kernel 2: per 16-row block: bf16 WMMA GEMM (input@w_i^T + h_prev@w_h^T)
//             into 256KB LDS, then fused layernorms + gates + c/h epilogue.
// ---------------------------------------------------------------------------

typedef __attribute__((ext_vector_type(16))) __bf16 v16bf;
typedef __attribute__((ext_vector_type(8)))  __bf16 v8bf;
typedef __attribute__((ext_vector_type(8)))  float  v8f;

#define BSZ   8192
#define KDIM  1024
#define NG    4096   // 4*H
#define HDIM  1024

// ---------------------------------------------------------------------------
// f32 -> bf16 conversion (8 elements / thread, 2x float4 load, 1x 16B store)
// ---------------------------------------------------------------------------
__global__ __launch_bounds__(256)
void cvt_f32_to_bf16(const float* __restrict__ s, __bf16* __restrict__ d, int n) {
    int i = (blockIdx.x * 256 + threadIdx.x) * 8;
    if (i >= n) return;
    float4 a = *(const float4*)(s + i);
    float4 b = *(const float4*)(s + i + 4);
    v8bf o;
    o[0] = (__bf16)a.x; o[1] = (__bf16)a.y; o[2] = (__bf16)a.z; o[3] = (__bf16)a.w;
    o[4] = (__bf16)b.x; o[5] = (__bf16)b.y; o[6] = (__bf16)b.z; o[7] = (__bf16)b.w;
    *(v8bf*)(d + i) = o;
}

// ---------------------------------------------------------------------------
// Fragment helpers (wave32 WMMA; A/B use this code's contiguous k order)
// ---------------------------------------------------------------------------
// A (16x32 bf16, MxK): lane = half*16 + m ; elements 0..15 -> K=half*16..+15
__device__ inline v16bf load_a_frag(const __bf16* __restrict__ rowp, int half) {
    v8bf lo = *(const v8bf*)(rowp + half * 16);
    v8bf hi = *(const v8bf*)(rowp + half * 16 + 8);
    v16bf r;
#pragma unroll
    for (int i = 0; i < 8; ++i) { r[i] = lo[i]; r[i + 8] = hi[i]; }
    return r;
}

// B (32x16 bf16, KxN): lane = half*16 + n ; elements 0..15 -> K=half*16..+15
// w stored [N,K] row-major, so column n of B == row n of w (K-contiguous).
__device__ inline v16bf load_b_frag(const __bf16* __restrict__ w,
                                    int col, int kt, int half) {
    return *(const v16bf*)(w + (size_t)col * KDIM + kt + half * 16);
}

// C/D (16x16 f32): element e, half -> row m = e + half*8, col n = lane&15
__device__ inline void store_c_tile(float* __restrict__ sm, v8f acc,
                                    int ncol, int half, int l15) {
    int n = ncol + l15;
#pragma unroll
    for (int e = 0; e < 8; ++e)
        sm[(size_t)(e + half * 8) * NG + n] = acc[e];
}

__device__ inline float wave_sum(float v) {
#pragma unroll
    for (int o = 16; o > 0; o >>= 1) v += __shfl_xor(v, o, 32);
    return v;
}

// In-place layernorm + activation over a 1024-wide LDS row segment.
// One wave32 per call; lane handles columns j = i*32 + lane (bank-conflict free).
template <bool TANH>
__device__ inline void ln_act(float* __restrict__ buf,
                              const float* __restrict__ g,
                              const float* __restrict__ b, int lane) {
    float x[32], s = 0.f, s2 = 0.f;
#pragma unroll
    for (int i = 0; i < 32; ++i) {
        float v = buf[i * 32 + lane];
        x[i] = v; s += v; s2 += v * v;
    }
    s = wave_sum(s); s2 = wave_sum(s2);
    float mu  = s * (1.f / 1024.f);
    float inv = rsqrtf(s2 * (1.f / 1024.f) - mu * mu + 1e-5f);
#pragma unroll
    for (int i = 0; i < 32; ++i) {
        int j = i * 32 + lane;
        float y = (x[i] - mu) * inv * g[j] + b[j];
        buf[j] = TANH ? tanhf(y) : 1.f / (1.f + __expf(-y));
    }
}

// ---------------------------------------------------------------------------
// Fused kernel: one block = 16 rows of the batch. 8 waves x 256 threads.
//   Phase 1: WMMA GEMM -> LDS gates[16][4096] (f32)
//   Phase 2: layernorms + gates + c/h epilogue (1 wave per 2 rows)
// ---------------------------------------------------------------------------
__global__ __launch_bounds__(256)
void lstm_fused(const __bf16* __restrict__ Abf,   // input  bf16 [B,K]
                const __bf16* __restrict__ Hbf,   // h_prev bf16 [B,K]
                const __bf16* __restrict__ Wibf,  // w_i    bf16 [4H,K]
                const __bf16* __restrict__ Whbf,  // w_h    bf16 [4H,K]
                const float* __restrict__ c_prev,
                const float* __restrict__ lni_g, const float* __restrict__ lni_b,
                const float* __restrict__ lnh_g, const float* __restrict__ lnh_b,
                const float* __restrict__ lng_g, const float* __restrict__ lng_b,
                const float* __restrict__ lno_g, const float* __restrict__ lno_b,
                const float* __restrict__ lnc_g, const float* __restrict__ lnc_b,
                float* __restrict__ out_h, float* __restrict__ out_c) {
    extern __shared__ float sm[];  // gates[16][4096] f32 = 256 KB LDS

    const int tid  = threadIdx.x;
    const int lane = tid & 31;
    const int w    = tid >> 5;        // wave id 0..7
    const int half = lane >> 4;
    const int l15  = lane & 15;
    const int r0   = blockIdx.x * 16; // first batch row of this block

    const __bf16* arow_i = Abf + (size_t)(r0 + l15) * KDIM;
    const __bf16* arow_h = Hbf + (size_t)(r0 + l15) * KDIM;

    // ---- Phase 1: GEMM. Each wave owns 8 groups of 4 N-tiles (64 cols). ----
    for (int ng = 0; ng < 8; ++ng) {
        const int nb = (w * 8 + ng) * 64;  // base gate column of this group
        // Faithful reference "bug": the g-chunk (cols 2048..3071) uses only the
        // h_prev @ w_h contribution (h_g); the input @ w_i chunk is dropped.
        const bool skip_wi = (nb >= 2048) && (nb < 3072);

        v8f acc0 = {}, acc1 = {}, acc2 = {}, acc3 = {};

        if (!skip_wi) {
#pragma unroll 4
            for (int kt = 0; kt < KDIM; kt += 32) {
                v16bf a  = load_a_frag(arow_i + kt, half);
                v16bf b0 = load_b_frag(Wibf, nb + 0,  kt, half);
                v16bf b1 = load_b_frag(Wibf, nb + 16, kt, half);
                v16bf b2 = load_b_frag(Wibf, nb + 32, kt, half);
                v16bf b3 = load_b_frag(Wibf, nb + 48, kt, half);
                acc0 = __builtin_amdgcn_wmma_f32_16x16x32_bf16(false, a, false, b0, (short)0, acc0, false, false);
                acc1 = __builtin_amdgcn_wmma_f32_16x16x32_bf16(false, a, false, b1, (short)0, acc1, false, false);
                acc2 = __builtin_amdgcn_wmma_f32_16x16x32_bf16(false, a, false, b2, (short)0, acc2, false, false);
                acc3 = __builtin_amdgcn_wmma_f32_16x16x32_bf16(false, a, false, b3, (short)0, acc3, false, false);
            }
        }
#pragma unroll 4
        for (int kt = 0; kt < KDIM; kt += 32) {
            v16bf a  = load_a_frag(arow_h + kt, half);
            v16bf b0 = load_b_frag(Whbf, nb + 0,  kt, half);
            v16bf b1 = load_b_frag(Whbf, nb + 16, kt, half);
            v16bf b2 = load_b_frag(Whbf, nb + 32, kt, half);
            v16bf b3 = load_b_frag(Whbf, nb + 48, kt, half);
            acc0 = __builtin_amdgcn_wmma_f32_16x16x32_bf16(false, a, false, b0, (short)0, acc0, false, false);
            acc1 = __builtin_amdgcn_wmma_f32_16x16x32_bf16(false, a, false, b1, (short)0, acc1, false, false);
            acc2 = __builtin_amdgcn_wmma_f32_16x16x32_bf16(false, a, false, b2, (short)0, acc2, false, false);
            acc3 = __builtin_amdgcn_wmma_f32_16x16x32_bf16(false, a, false, b3, (short)0, acc3, false, false);
        }

        store_c_tile(sm, acc0, nb + 0,  half, l15);
        store_c_tile(sm, acc1, nb + 16, half, l15);
        store_c_tile(sm, acc2, nb + 32, half, l15);
        store_c_tile(sm, acc3, nb + 48, half, l15);
    }

    __syncthreads();

    // ---- Phase 2: LN + LSTM epilogue. Wave w handles rows 2w, 2w+1. ----
#pragma unroll
    for (int rr = 0; rr < 2; ++rr) {
        const int r    = w * 2 + rr;
        const size_t grow = (size_t)(r0 + r);
        float* gi_ = sm + (size_t)r * NG;        // gate i  (gi+gh chunk 0)
        float* gf_ = gi_ + 1024;                 // gate f  (gi+gh chunk 1)
        float* gg_ = gi_ + 2048;                 // h_g only (chunk 2, h part)
        float* go_ = gi_ + 3072;                 // gate o  (gi+gh chunk 3)

        ln_act<false>(gi_, lni_g, lni_b, lane);  // i_g = sigmoid(LN(i_i+h_i))
        ln_act<false>(gf_, lnh_g, lnh_b, lane);  // f_g = sigmoid(LN(i_f+h_f))
        ln_act<false>(go_, lno_g, lno_b, lane);  // o_g = sigmoid(LN(i_o+h_o))

        // g pre-activation = i_g + h_g   (faithful to the source bug)
#pragma unroll
        for (int i = 0; i < 32; ++i) {
            int j = i * 32 + lane;
            gg_[j] += gi_[j];
        }
        ln_act<true>(gg_, lng_g, lng_b, lane);   // g_g = tanh(LN(i_g + h_g))

        // c_new = LN(f_g*c_prev + i_g*g_g) ; h_new = o_g * tanh(c_new)
        float x[32], s = 0.f, s2 = 0.f;
#pragma unroll
        for (int i = 0; i < 32; ++i) {
            int j = i * 32 + lane;
            float v = gf_[j] * c_prev[grow * HDIM + j] + gi_[j] * gg_[j];
            x[i] = v; s += v; s2 += v * v;
        }
        s = wave_sum(s); s2 = wave_sum(s2);
        float mu  = s * (1.f / 1024.f);
        float inv = rsqrtf(s2 * (1.f / 1024.f) - mu * mu + 1e-5f);
#pragma unroll
        for (int i = 0; i < 32; ++i) {
            int j = i * 32 + lane;
            float cn = (x[i] - mu) * inv * lnc_g[j] + lnc_b[j];
            out_c[grow * HDIM + j] = cn;
            out_h[grow * HDIM + j] = go_[j] * tanhf(cn);
        }
    }
}

// ---------------------------------------------------------------------------
// Launcher
// ---------------------------------------------------------------------------
extern "C" void kernel_launch(void* const* d_in, const int* in_sizes, int n_in,
                              void* d_out, int out_size, void* d_ws, size_t ws_size,
                              hipStream_t stream) {
    (void)in_sizes; (void)n_in; (void)out_size; (void)ws_size;

    const float* input  = (const float*)d_in[0];
    const float* h_prev = (const float*)d_in[1];
    const float* c_prev = (const float*)d_in[2];
    const float* w_i    = (const float*)d_in[3];
    const float* w_h    = (const float*)d_in[4];
    const float* lni_g  = (const float*)d_in[5];
    const float* lni_b  = (const float*)d_in[6];
    const float* lnh_g  = (const float*)d_in[7];
    const float* lnh_b  = (const float*)d_in[8];
    const float* lng_g  = (const float*)d_in[9];
    const float* lng_b  = (const float*)d_in[10];
    const float* lno_g  = (const float*)d_in[11];
    const float* lno_b  = (const float*)d_in[12];
    const float* lnc_g  = (const float*)d_in[13];
    const float* lnc_b  = (const float*)d_in[14];

    // workspace layout (48 MB): bf16 copies of A, H, Wi, Wh
    char* ws = (char*)d_ws;
    __bf16* Abf  = (__bf16*)(ws);                          // 16 MB
    __bf16* Hbf  = (__bf16*)(ws + ((size_t)16 << 20));     // 16 MB
    __bf16* Wibf = (__bf16*)(ws + ((size_t)32 << 20));     //  8 MB
    __bf16* Whbf = (__bf16*)(ws + ((size_t)40 << 20));     //  8 MB

    const int nAct = BSZ * KDIM;     // 8M elements each
    const int nW   = NG * KDIM;      // 4M elements each
    cvt_f32_to_bf16<<<nAct / (256 * 8), 256, 0, stream>>>(input,  Abf,  nAct);
    cvt_f32_to_bf16<<<nAct / (256 * 8), 256, 0, stream>>>(h_prev, Hbf,  nAct);
    cvt_f32_to_bf16<<<nW   / (256 * 8), 256, 0, stream>>>(w_i,    Wibf, nW);
    cvt_f32_to_bf16<<<nW   / (256 * 8), 256, 0, stream>>>(w_h,    Whbf, nW);

    float* out_h = (float*)d_out;
    float* out_c = out_h + (size_t)BSZ * HDIM;

    // 512 blocks x 256 threads, 256 KB dynamic LDS (<= 320 KB/WGP on CDNA5)
    lstm_fused<<<BSZ / 16, 256, 16 * NG * sizeof(float), stream>>>(
        Abf, Hbf, Wibf, Whbf, c_prev,
        lni_g, lni_b, lnh_g, lnh_b, lng_g, lng_b, lno_g, lno_b, lnc_g, lnc_b,
        out_h, out_c);
}
